// MaskedMHA_64570538328540
// MI455X (gfx1250) — matrix operand order repaired and checked by reading, not compiled, not run on hardware
//
#include <hip/hip_runtime.h>
#include <hip/hip_bf16.h>

// ---------------------------------------------------------------------------
// MaskedMHA on gfx1250 (MI455X): bf16 WMMA everywhere, f32 accumulate.
// GEMM B-operand staged to LDS with async global->LDS DMA (ASYNCcnt).
// ---------------------------------------------------------------------------

typedef __bf16 bf16;
typedef __attribute__((ext_vector_type(8)))  bf16  bf16x8;
typedef __attribute__((ext_vector_type(16))) bf16  bf16x16;
typedef __attribute__((ext_vector_type(8)))  float f32x8;

#define EMBD  1024
#define HEADS 16
#define DHEAD 64
#define TQ    2048
#define TH    1024
#define TK    (TQ + TH)       // 3072
#define SCALE_QK 0.3535533905932738f  // 64^-0.25
#define NEGINF  (-1e9f)

__device__ __forceinline__ f32x8 wmma_bf16(bf16x16 a, bf16x16 b, f32x8 c) {
  // D = A(16x32) * B(32x16) + C, f32 accumulate
  return __builtin_amdgcn_wmma_f32_16x16x32_bf16(
      /*neg_a=*/false, a, /*neg_b=*/false, b,
      /*c_mod=*/(short)0, c, /*reuse_a=*/false, /*reuse_b=*/false);
}

// A-fragment (16x32, row in lane%16): elements 0..15 = K(off..off+15),
// off = (lane<16 ? 0 : 16).
__device__ __forceinline__ bf16x16 load_a_frag(const bf16* rowk, int lane) {
  const int off = (lane < 16) ? 0 : 16;
  bf16x8 lo = *(const bf16x8*)(rowk + off);
  bf16x8 hi = *(const bf16x8*)(rowk + off + 8);
  return __builtin_shufflevector(lo, hi, 0,1,2,3,4,5,6,7,8,9,10,11,12,13,14,15);
}

// B-fragment from 16B-aligned storage (LDS): elements 0..15 = K(base..base+15)
__device__ __forceinline__ bf16x16 load_b_frag16(const bf16* colk, int lane) {
  const int off = (lane < 16) ? 0 : 16;
  bf16x8 lo = *(const bf16x8*)(colk + off);
  bf16x8 hi = *(const bf16x8*)(colk + off + 8);
  return __builtin_shufflevector(lo, hi, 0,1,2,3,4,5,6,7,8,9,10,11,12,13,14,15);
}

// Async global -> LDS copy of 16 bytes (ASYNCcnt-tracked DMA path).
// lds_off: byte offset into the wave's LDS allocation (low 32b of generic ptr).
__device__ __forceinline__ void async_copy_b128(unsigned int lds_off,
                                                const bf16* gaddr) {
  asm volatile("global_load_async_to_lds_b128 %0, %1, off"
               :: "v"(lds_off), "v"(gaddr)
               : "memory");
}

// ---------------------------------------------------------------------------
// Elementwise f32 -> bf16
// ---------------------------------------------------------------------------
__global__ void cvt_bf16_kernel(const float* __restrict__ src,
                                bf16* __restrict__ dst, int n) {
  for (int i = blockIdx.x * blockDim.x + threadIdx.x; i < n;
       i += gridDim.x * blockDim.x)
    dst[i] = (bf16)src[i];
}

// ---------------------------------------------------------------------------
// Transposed convert:  src[b][C][Tsrc] f32  ->  dst[b][TK][C] bf16 (rows t)
// ---------------------------------------------------------------------------
__global__ void transpose_bf16_kernel(const float* __restrict__ src,
                                      bf16* __restrict__ dst,
                                      int Tsrc, int tOff) {
  __shared__ float tile[32][33];
  const int t0 = blockIdx.x * 32, c0 = blockIdx.y * 32, b = blockIdx.z;
  const float* sb = src + (size_t)b * EMBD * Tsrc;
  bf16* db = dst + (size_t)b * TK * EMBD;
  #pragma unroll
  for (int r = threadIdx.y; r < 32; r += 8)
    tile[r][threadIdx.x] = sb[(size_t)(c0 + r) * Tsrc + t0 + threadIdx.x];
  __syncthreads();
  #pragma unroll
  for (int r = threadIdx.y; r < 32; r += 8)
    db[(size_t)(tOff + t0 + r) * EMBD + c0 + threadIdx.x] =
        (bf16)tile[threadIdx.x][r];
}

// ---------------------------------------------------------------------------
// Concatenated mask flags: flag[b][tk] = 1.0 valid / 0.0 masked
// ---------------------------------------------------------------------------
__global__ void build_flag_kernel(const unsigned char* __restrict__ mask,
                                  const unsigned char* __restrict__ hmask,
                                  float* __restrict__ flag) {
  int i = blockIdx.x * blockDim.x + threadIdx.x;
  if (i >= 2 * TK) return;
  int b = i / TK, t = i % TK;
  unsigned char m = (t < TQ) ? mask[b * TQ + t] : hmask[b * TH + (t - TQ)];
  flag[i] = m ? 1.0f : 0.0f;
}

// ---------------------------------------------------------------------------
// WMMA GEMM:  Y[b] = A(MxK) * Bt(NxK)^T + bias, 128x128 block, 8 waves.
// B^T K-slab (128 x 32 bf16) double-buffered in LDS via async global->LDS.
//   MODE 0: store bf16 transposed  Yt[n][m], scaled   (Q^T / K^T)
//   MODE 1: store bf16 natural     Y[m][n]            (V)
//   MODE 2: store f32  natural     Y[m][n]            (final out)
// ---------------------------------------------------------------------------
#define BPITCH 40   // LDS row pitch (bf16 elems): 80B, conflict-free & 16B-aligned

template <int MODE>
__global__ __launch_bounds__(256)
void gemm_wmma_kernel(const bf16* __restrict__ A, const bf16* __restrict__ Bt,
                      const float* __restrict__ bias, void* __restrict__ Yv,
                      int M, int N, int K, long long strideBt, long long strideY,
                      float scale) {
  __shared__ bf16 stage[2][128][BPITCH];

  const int b = blockIdx.z;
  const bf16* Btb = Bt + (size_t)b * strideBt;
  const int lane = threadIdx.x & 31, w = threadIdx.x >> 5;
  const int ln16 = lane & 15;
  const int wm = (w & 1) * 64, wn = (w >> 1) * 32;
  const int blockM = blockIdx.y * 128, blockN = blockIdx.x * 128;
  const int offB = (lane < 16) ? 0 : 16;

  // ---- async staging assignment: thread -> (row, 32B strip) ----
  const int ct = threadIdx.x;
  const int crow = ct >> 1, cpart = ct & 1;          // 128 rows x 2 strips
  const bf16* gsrc = Btb + (size_t)(blockN + crow) * K + cpart * 16;
  const unsigned int lds0 =
      (unsigned int)(size_t)&stage[0][crow][cpart * 16];
  const unsigned int ldsStride =
      (unsigned int)((size_t)&stage[1][0][0] - (size_t)&stage[0][0][0]);

  f32x8 acc[4][2];
  #pragma unroll
  for (int mi = 0; mi < 4; ++mi)
    #pragma unroll
    for (int ni = 0; ni < 2; ++ni)
      acc[mi][ni] = f32x8{0,0,0,0,0,0,0,0};

  const bf16* Arow[4];
  #pragma unroll
  for (int mi = 0; mi < 4; ++mi)
    Arow[mi] = A + (size_t)(blockM + wm + mi * 16 + ln16) * K;

  // prologue: stage first K-slab
  {
    unsigned int l = lds0;
    async_copy_b128(l, gsrc);
    async_copy_b128(l + 16, gsrc + 8);
  }

  int cur = 0;
  for (int kb = 0; kb < K; kb += 32) {
    const bool more = (kb + 32) < K;
    if (more) {  // issue DMA for next slab into the other buffer
      unsigned int l = lds0 + (1 - cur) * ldsStride;
      const bf16* g = gsrc + (kb + 32);
      async_copy_b128(l, g);
      async_copy_b128(l + 16, g + 8);
      asm volatile("s_wait_asynccnt 0x2" ::: "memory");
    } else {
      asm volatile("s_wait_asynccnt 0x0" ::: "memory");
    }
    __syncthreads();   // current slab visible to all waves

    bf16x16 af[4], bfr[2];
    #pragma unroll
    for (int mi = 0; mi < 4; ++mi) {
      af[mi] = load_a_frag(Arow[mi] + kb, lane);
      __builtin_prefetch(Arow[mi] + kb + 64, 0, 0);   // global_prefetch next A
    }
    #pragma unroll
    for (int ni = 0; ni < 2; ++ni)
      bfr[ni] = load_b_frag16(&stage[cur][wn + ni * 16 + ln16][0], lane);
    #pragma unroll
    for (int mi = 0; mi < 4; ++mi)
      #pragma unroll
      for (int ni = 0; ni < 2; ++ni)
        acc[mi][ni] = wmma_bf16(af[mi], bfr[ni], acc[mi][ni]);

    __syncthreads();   // all reads done before this buffer is re-filled
    cur ^= 1;
  }

  const int hh = (lane >= 16) ? 8 : 0;
  #pragma unroll
  for (int mi = 0; mi < 4; ++mi) {
    const int gm0 = blockM + wm + mi * 16 + hh;
    #pragma unroll
    for (int ni = 0; ni < 2; ++ni) {
      const int gn = blockN + wn + ni * 16 + ln16;
      if (MODE == 0) {
        bf16* Y = (bf16*)Yv + (size_t)b * strideY;
        bf16x8 o;
        #pragma unroll
        for (int i = 0; i < 8; ++i)
          o[i] = (bf16)((acc[mi][ni][i] + bias[gm0 + i]) * scale);
        *(bf16x8*)(Y + (size_t)gn * M + gm0) = o;   // Yt[n][m], 16B store
      } else if (MODE == 1) {
        bf16* Y = (bf16*)Yv + (size_t)b * strideY;
        #pragma unroll
        for (int i = 0; i < 8; ++i)
          Y[(size_t)(gm0 + i) * N + gn] =
              (bf16)((acc[mi][ni][i] + bias[gm0 + i]) * scale);
      } else {
        float* Y = (float*)Yv + (size_t)b * strideY;
        #pragma unroll
        for (int i = 0; i < 8; ++i)
          Y[(size_t)(gm0 + i) * N + gn] = acc[mi][ni][i] + bias[gm0 + i];
      }
    }
  }
}

// ---------------------------------------------------------------------------
// Flash attention: qt[b][Tq][E] (pre-scaled), kt[b][Tk][E] (pre-scaled),
// v[b][E][Tk], flag[b][Tk]  ->  ot[b][Tq][E] bf16.
// 4 waves / block, each wave: 16 q-rows, streams Tk in 64-wide chunks.
// ---------------------------------------------------------------------------
__global__ __launch_bounds__(128)
void flash_attn_kernel(const bf16* __restrict__ qt, const bf16* __restrict__ kt,
                       const bf16* __restrict__ vmat,
                       const float* __restrict__ flag, bf16* __restrict__ ot) {
  const int b = blockIdx.z, h = blockIdx.y;
  const int lane = threadIdx.x & 31, w = threadIdx.x >> 5;
  const int ln16 = lane & 15;
  const int hh = (lane >= 16) ? 8 : 0;
  const int offB = (lane < 16) ? 0 : 16;
  const int q0 = blockIdx.x * 64 + w * 16;
  const int hbase = h * DHEAD;

  const bf16* qtb = qt + (size_t)b * TQ * EMBD;
  const bf16* ktb = kt + (size_t)b * TK * EMBD;
  const bf16* vb  = vmat + (size_t)b * EMBD * TK;
  const float* fl = flag + (size_t)b * TK;

  __shared__ bf16 Pl[4][16][72];   // per-wave P tile (16 x 64, padded pitch)

  // Q fragments (16 rows x d=64) -- reused for the whole Tk sweep.
  bf16x16 qf[2];
  {
    const bf16* qrow = qtb + (size_t)(q0 + ln16) * EMBD + hbase;
    #pragma unroll
    for (int kk = 0; kk < 2; ++kk) qf[kk] = load_a_frag(qrow + kk * 32, lane);
  }

  f32x8 o[4];
  #pragma unroll
  for (int jd = 0; jd < 4; ++jd) o[jd] = f32x8{0,0,0,0,0,0,0,0};
  float mrun[8], lrun[8];
  #pragma unroll
  for (int i = 0; i < 8; ++i) { mrun[i] = -3e38f; lrun[i] = 0.0f; }

  for (int tk0 = 0; tk0 < TK; tk0 += 64) {
    // ---- S = (q*s)(k*s)^T for 16x64 tile, masked ----
    f32x8 s[4];
    #pragma unroll
    for (int jn = 0; jn < 4; ++jn) {
      f32x8 a = f32x8{0,0,0,0,0,0,0,0};
      const bf16* kcol = ktb + (size_t)(tk0 + jn * 16 + ln16) * EMBD + hbase;
      #pragma unroll
      for (int kk = 0; kk < 2; ++kk) {
        bf16x16 bfr = *(const bf16x16*)(kcol + kk * 32 + offB);
        a = wmma_bf16(qf[kk], bfr, a);
      }
      const float keep = fl[tk0 + jn * 16 + ln16];  // column-constant
      #pragma unroll
      for (int i = 0; i < 8; ++i) s[jn][i] = (keep != 0.0f) ? a[i] : NEGINF;
    }

    // ---- online softmax (rows live across 16-lane halves) ----
    float alpha[8];
    #pragma unroll
    for (int i = 0; i < 8; ++i) {
      float v = fmaxf(fmaxf(s[0][i], s[1][i]), fmaxf(s[2][i], s[3][i]));
      v = fmaxf(v, __shfl_xor(v, 1));
      v = fmaxf(v, __shfl_xor(v, 2));
      v = fmaxf(v, __shfl_xor(v, 4));
      v = fmaxf(v, __shfl_xor(v, 8));
      const float mn = fmaxf(mrun[i], v);
      alpha[i] = __expf(mrun[i] - mn);
      mrun[i] = mn;
    }
    #pragma unroll
    for (int jn = 0; jn < 4; ++jn)
      #pragma unroll
      for (int i = 0; i < 8; ++i) s[jn][i] = __expf(s[jn][i] - mrun[i]);
    #pragma unroll
    for (int i = 0; i < 8; ++i) {
      float r = (s[0][i] + s[1][i]) + (s[2][i] + s[3][i]);
      r += __shfl_xor(r, 1);
      r += __shfl_xor(r, 2);
      r += __shfl_xor(r, 4);
      r += __shfl_xor(r, 8);
      lrun[i] = lrun[i] * alpha[i] + r;
    }
    #pragma unroll
    for (int jd = 0; jd < 4; ++jd)
      #pragma unroll
      for (int i = 0; i < 8; ++i) o[jd][i] *= alpha[i];

    // ---- restripe P (C-layout -> A-layout) through per-wave LDS ----
    #pragma unroll
    for (int jn = 0; jn < 4; ++jn)
      #pragma unroll
      for (int i = 0; i < 8; ++i)
        Pl[w][i + hh][jn * 16 + ln16] = (bf16)s[jn][i];
    __syncthreads();

    // ---- O += P * V ----
    #pragma unroll
    for (int jd = 0; jd < 4; ++jd) {
      const bf16* vrow = vb + (size_t)(hbase + jd * 16 + ln16) * TK + tk0;
      #pragma unroll
      for (int kk = 0; kk < 2; ++kk) {
        bf16x16 pf = load_a_frag(&Pl[w][ln16][kk * 32], lane);
        bf16x16 vf = *(const bf16x16*)(vrow + kk * 32 + offB);
        o[jd] = wmma_bf16(pf, vf, o[jd]);
      }
    }
    __syncthreads();
  }

  // ---- epilogue: O / l ->  ot[b][t][e] bf16 ----
  #pragma unroll
  for (int jd = 0; jd < 4; ++jd) {
    const int e = hbase + jd * 16 + ln16;
    #pragma unroll
    for (int i = 0; i < 8; ++i) {
      const int t = q0 + i + hh;
      ot[((size_t)b * TQ + t) * EMBD + e] = (bf16)(o[jd][i] / lrun[i]);
    }
  }
}

// ---------------------------------------------------------------------------
extern "C" void kernel_launch(void* const* d_in, const int* in_sizes, int n_in,
                              void* d_out, int out_size, void* d_ws,
                              size_t ws_size, hipStream_t stream) {
  const float*         x       = (const float*)d_in[0];
  const unsigned char* mask    = (const unsigned char*)d_in[1];
  const float*         history = (const float*)d_in[2];
  const unsigned char* hmask   = (const unsigned char*)d_in[3];
  const float* Wq = (const float*)d_in[4];  const float* bq = (const float*)d_in[5];
  const float* Wk = (const float*)d_in[6];  const float* bk = (const float*)d_in[7];
  const float* Wv = (const float*)d_in[8];  const float* bv = (const float*)d_in[9];
  const float* Wp = (const float*)d_in[10]; const float* bp = (const float*)d_in[11];

  char* ws = (char*)d_ws;
  size_t off = 0;
  auto alloc = [&](size_t bytes) {
    char* p = ws + off;
    off = (off + bytes + 255) & ~(size_t)255;
    return p;
  };
  const size_t WSZ = (size_t)EMBD * EMBD;  // 1M elements
  bf16* wq_bf = (bf16*)alloc(WSZ * 2);
  bf16* wk_bf = (bf16*)alloc(WSZ * 2);
  bf16* wv_bf = (bf16*)alloc(WSZ * 2);
  bf16* wp_bf = (bf16*)alloc(WSZ * 2);
  bf16* xcat  = (bf16*)alloc((size_t)2 * TK * EMBD * 2);  // X^T || H^T, bf16
  bf16* qtw   = (bf16*)alloc((size_t)2 * TQ * EMBD * 2);  // Q^T, pre-scaled
  bf16* ktw   = (bf16*)alloc((size_t)2 * TK * EMBD * 2);  // K^T, pre-scaled
  bf16* vw    = (bf16*)alloc((size_t)2 * EMBD * TK * 2);  // V natural
  bf16* otw   = (bf16*)alloc((size_t)2 * TQ * EMBD * 2);  // attention out^T
  float* flag = (float*)alloc((size_t)2 * TK * 4);

  // 1) weights -> bf16
  cvt_bf16_kernel<<<4096, 256, 0, stream>>>(Wq, wq_bf, (int)WSZ);
  cvt_bf16_kernel<<<4096, 256, 0, stream>>>(Wk, wk_bf, (int)WSZ);
  cvt_bf16_kernel<<<4096, 256, 0, stream>>>(Wv, wv_bf, (int)WSZ);
  cvt_bf16_kernel<<<4096, 256, 0, stream>>>(Wp, wp_bf, (int)WSZ);

  // 2) x / history -> transposed bf16 concat [b][TK][C]
  transpose_bf16_kernel<<<dim3(TQ / 32, EMBD / 32, 2), dim3(32, 8), 0, stream>>>(
      x, xcat, TQ, 0);
  transpose_bf16_kernel<<<dim3(TH / 32, EMBD / 32, 2), dim3(32, 8), 0, stream>>>(
      history, xcat, TH, TQ);

  // 3) mask flags
  build_flag_kernel<<<(2 * TK + 255) / 256, 256, 0, stream>>>(mask, hmask, flag);

  // 4) projections (WMMA GEMMs, async-staged B)
  gemm_wmma_kernel<0><<<dim3(TQ / 128, EMBD / 128, 2), 256, 0, stream>>>(
      wq_bf, xcat, bq, qtw, EMBD, TQ, EMBD,
      (long long)TK * EMBD, (long long)TQ * EMBD, SCALE_QK);
  gemm_wmma_kernel<0><<<dim3(TK / 128, EMBD / 128, 2), 256, 0, stream>>>(
      wk_bf, xcat, bk, ktw, EMBD, TK, EMBD,
      (long long)TK * EMBD, (long long)TK * EMBD, SCALE_QK);
  gemm_wmma_kernel<1><<<dim3(TK / 128, EMBD / 128, 2), 256, 0, stream>>>(
      wv_bf, xcat, bv, vw, EMBD, TK, EMBD,
      (long long)TK * EMBD, (long long)EMBD * TK, 1.0f);

  // 5) flash attention
  flash_attn_kernel<<<dim3(TQ / 64, HEADS, 2), 128, 0, stream>>>(
      qtw, ktw, vw, flag, otw);

  // 6) output projection -> f32 d_out [b][C][Tq]
  gemm_wmma_kernel<2><<<dim3(TQ / 128, EMBD / 128, 2), 256, 0, stream>>>(
      wp_bf, otw, bp, d_out, EMBD, TQ, EMBD,
      (long long)TQ * EMBD, (long long)EMBD * TQ, 1.0f);
}
